// LSTMModel_74947179316253
// MI455X (gfx1250) — matrix operand order repaired and checked by reading, not compiled, not run on hardware
//
#include <hip/hip_runtime.h>
#include <hip/hip_bf16.h>
#include <stdint.h>

// ---------------------------------------------------------------------------
// 2-layer LSTM for MI455X (gfx1250), wave32 + WMMA bf16.
//   B=16, T=512, F=1024, gates G=4F=4096.
//   1) Xg[t] = x_t @ Wi + b precomputed as one parallel WMMA GEMM (bf16 out).
//   2) Persistent recurrent kernel: 256 waves == 256 N-tiles of the 4096-wide
//      gate matrix; per step 32x v_wmma_f32_16x16x32_bf16 over K=1024 against
//      L2-resident Wh (8 MB bf16), grid barrier, pointwise cell, barrier.
//   GEMM inner loops are software-pipelined (prefetch next K-chunk fragments
//   before issuing current WMMAs) so loads overlap the WMMA pipe.
// ---------------------------------------------------------------------------

typedef __attribute__((ext_vector_type(16))) __bf16 v16bf;
typedef __attribute__((ext_vector_type(8)))  float  v8f;

#define DEVFN static __device__ __forceinline__

static constexpr int T_    = 512;
static constexpr int B_    = 16;
static constexpr int F_    = 1024;
static constexpr int G_    = 4096;          // 4*F
static constexpr int NT_   = G_ / 16;       // 256 N-tiles
static constexpr int KC_   = F_ / 32;       // 32 K-chunks
static constexpr int RBLK_ = 32;            // persistent blocks (256 waves)

DEVFN unsigned short f32_to_bf16(float f) {
  unsigned u = __float_as_uint(f);
  u += 0x7FFFu + ((u >> 16) & 1u);          // round-to-nearest-even
  return (unsigned short)(u >> 16);
}
DEVFN float bf16_to_f32(unsigned short h) {
  return __uint_as_float(((unsigned)h) << 16);
}
DEVFN unsigned pack2(float lo, float hi) {
  return (unsigned)f32_to_bf16(lo) | ((unsigned)f32_to_bf16(hi) << 16);
}
DEVFN float sigmoidf(float x) { return 1.0f / (1.0f + expf(-x)); }

DEVFN v16bf wmma_bf16(v16bf a, v16bf b, v8f& acc);  // fwd decl not needed; inline below

// A-fragment (16x32 bf16) per-lane layout index for element (m, k%32==kk):
//   lane = m + 16*half, half = (kk>>3)&1, vgpr = (kk>>4)*4 + ((kk&7)>>1), pos = kk&1
DEVFN int hpack_ushort_index(int m, int j) {   // j = k in [0,F)
  int chunk = j >> 5, kk = j & 31;
  int sub  = kk >> 3;
  int half = sub & 1;
  int v    = (sub >> 1) * 4 + ((kk & 7) >> 1);
  int pos  = kk & 1;
  int lane = m + 16 * half;
  return (((chunk * 32) + lane) * 8 + v) * 2 + pos;   // ushort index
}

// ------------------------- pack kernels ------------------------------------

// x [B,T,F] f32 -> A-packed bf16 keyed [t][chunk][lane][8 dwords]
__global__ void pack_x_kernel(const float* __restrict__ x, unsigned* __restrict__ xp) {
  unsigned idx = blockIdx.x * 256u + threadIdx.x;     // dword idx, 4,194,304 total
  int v = idx & 7, lane = (idx >> 3) & 31, chunk = (idx >> 8) & 31, t = idx >> 13;
  int m = lane & 15, half = lane >> 4;
  int kk = (v >> 2) * 16 + half * 8 + (v & 3) * 2;
  int k  = chunk * 32 + kk;
  const float* s = x + ((size_t)m * T_ + t) * F_ + k; // x[b=m][t][k], k,k+1 contiguous
  xp[idx] = pack2(s[0], s[1]);
}

// W [F,G] f32 -> B-packed bf16 keyed [ntile][chunk][lane][8 dwords]
// B-fragment (32x16): lane = n + 16*halfB ; vgpr v holds k = chunk*32 + (v>>2)*16 + halfB*8 + 2*(v&3) (+pos)
__global__ void pack_w_kernel(const float* __restrict__ w, unsigned* __restrict__ wp) {
  unsigned idx = blockIdx.x * 256u + threadIdx.x;     // 2,097,152 dwords
  int v = idx & 7, lane = (idx >> 3) & 31, chunk = (idx >> 8) & 31, ntile = idx >> 13;
  int n = lane & 15, half = lane >> 4;
  int k = chunk * 32 + (v >> 2) * 16 + half * 8 + (v & 3) * 2;
  const float* s = w + (size_t)k * G_ + ntile * 16 + n;
  wp[idx] = pack2(s[0], s[G_]);                       // (k,n) and (k+1,n)
}

// init recurrent state: cstate = c0, hpackA = pack(h0), zero barrier vars
__global__ void init_state_kernel(const float* __restrict__ c0, const float* __restrict__ h0,
                                  float* __restrict__ cstate, unsigned* __restrict__ hpackA,
                                  unsigned* __restrict__ bar_cnt, unsigned* __restrict__ bar_gen) {
  int e = blockIdx.x * 256 + threadIdx.x;             // 16384 threads
  if (e == 0) { *bar_cnt = 0u; *bar_gen = 0u; }
  int m = e >> 10, j = e & 1023;
  cstate[e] = c0[e];
  ((unsigned short*)hpackA)[hpack_ushort_index(m, j)] = f32_to_bf16(h0[e]);
}

// ------------------------- Xg GEMM (parallel, no recurrence) ----------------
// Xg[t][m][n] (bf16) = sum_k apack(t) * wipack + bias[n]
// One wave -> (m_tile=t, 4 consecutive n_tiles).  32768 waves = 4096 blocks.
__global__ void gemm_xg_kernel(const unsigned* __restrict__ apack,
                               const unsigned* __restrict__ wpack,
                               const float* __restrict__ bias,
                               unsigned short* __restrict__ xg) {
  int w    = (blockIdx.x * 256 + threadIdx.x) >> 5;   // global wave id
  int lane = threadIdx.x & 31;
  int mt = w >> 6;                                    // t : 0..511
  int ng = (w & 63) * 4;                              // first of 4 n_tiles

  v8f a0 = {}, a1 = {}, a2 = {}, a3 = {};
  const unsigned* ap = apack + (size_t)mt * (KC_ * 256) + lane * 8;
  const unsigned* b0 = wpack + ((size_t)(ng + 0) * KC_ * 32 + lane) * 8;
  const unsigned* b1 = wpack + ((size_t)(ng + 1) * KC_ * 32 + lane) * 8;
  const unsigned* b2 = wpack + ((size_t)(ng + 2) * KC_ * 32 + lane) * 8;
  const unsigned* b3 = wpack + ((size_t)(ng + 3) * KC_ * 32 + lane) * 8;

  // software pipeline: fragments for chunk kc loaded one iteration ahead
  v16bf A  = *(const v16bf*)ap;
  v16bf B0 = *(const v16bf*)b0;
  v16bf B1 = *(const v16bf*)b1;
  v16bf B2 = *(const v16bf*)b2;
  v16bf B3 = *(const v16bf*)b3;

#pragma unroll 2
  for (int kc = 0; kc < KC_ - 1; ++kc) {
    ap += 256; b0 += 256; b1 += 256; b2 += 256; b3 += 256;
    v16bf An  = *(const v16bf*)ap;
    v16bf Bn0 = *(const v16bf*)b0;
    v16bf Bn1 = *(const v16bf*)b1;
    v16bf Bn2 = *(const v16bf*)b2;
    v16bf Bn3 = *(const v16bf*)b3;
    a0 = __builtin_amdgcn_wmma_f32_16x16x32_bf16(false, A, false, B0, (short)0, a0, false, false);
    a1 = __builtin_amdgcn_wmma_f32_16x16x32_bf16(false, A, false, B1, (short)0, a1, false, false);
    a2 = __builtin_amdgcn_wmma_f32_16x16x32_bf16(false, A, false, B2, (short)0, a2, false, false);
    a3 = __builtin_amdgcn_wmma_f32_16x16x32_bf16(false, A, false, B3, (short)0, a3, false, false);
    A = An; B0 = Bn0; B1 = Bn1; B2 = Bn2; B3 = Bn3;
  }
  a0 = __builtin_amdgcn_wmma_f32_16x16x32_bf16(false, A, false, B0, (short)0, a0, false, false);
  a1 = __builtin_amdgcn_wmma_f32_16x16x32_bf16(false, A, false, B1, (short)0, a1, false, false);
  a2 = __builtin_amdgcn_wmma_f32_16x16x32_bf16(false, A, false, B2, (short)0, a2, false, false);
  a3 = __builtin_amdgcn_wmma_f32_16x16x32_bf16(false, A, false, B3, (short)0, a3, false, false);

  int mh = lane >> 4, n = lane & 15;
  v8f accs[4] = {a0, a1, a2, a3};
#pragma unroll
  for (int ti = 0; ti < 4; ++ti) {
    int nb = (ng + ti) * 16;
    float bv = bias[nb + n];
#pragma unroll
    for (int r = 0; r < 8; ++r) {
      int m = r + 8 * mh;                              // C/D layout: VGPR r, lane half
      xg[((size_t)mt * 16 + m) * G_ + nb + n] = f32_to_bf16(accs[ti][r] + bv);
    }
  }
}

// ------------------------- grid barrier -------------------------------------
DEVFN void grid_sync(unsigned* cnt, unsigned* gen, unsigned nb, unsigned* lgen) {
  __syncthreads();
  if (threadIdx.x == 0) {
    unsigned next = *lgen + 1u;
    __threadfence();
    unsigned arrived = __hip_atomic_fetch_add(cnt, 1u, __ATOMIC_ACQ_REL, __HIP_MEMORY_SCOPE_AGENT);
    if (arrived == nb - 1u) {
      *cnt = 0u;
      __threadfence();
      __hip_atomic_store(gen, next, __ATOMIC_RELEASE, __HIP_MEMORY_SCOPE_AGENT);
    } else {
      while (__hip_atomic_load(gen, __ATOMIC_ACQUIRE, __HIP_MEMORY_SCOPE_AGENT) != next)
        __builtin_amdgcn_s_sleep(1);
    }
    *lgen = next;
    __threadfence();
  }
  __syncthreads();
}

// ------------------------- persistent recurrent kernel ----------------------
// 32 blocks x 256 threads = 256 waves; wave w owns gate N-tile w.
__global__ void lstm_recur_kernel(const unsigned* __restrict__ whpack,
                                  const unsigned short* __restrict__ xg,
                                  float* __restrict__ gates,
                                  unsigned* __restrict__ hpackA, unsigned* __restrict__ hpackB,
                                  float* __restrict__ cstate,
                                  unsigned* bar_cnt, unsigned* bar_gen,
                                  unsigned short* __restrict__ ypack,  // layer0: packed h-seq
                                  float* __restrict__ out_f32,         // layer1: d_out [B,T,F]
                                  float* __restrict__ cfin, float* __restrict__ hfin) {
  __shared__ unsigned lgen;
  if (threadIdx.x == 0) lgen = 0u;
  __syncthreads();

  int tid  = threadIdx.x;
  int gtid = blockIdx.x * 256 + tid;
  int wv   = gtid >> 5;                 // n_tile, 0..255
  int lane = tid & 31;
  const unsigned* bbase = whpack + ((size_t)wv * KC_ * 32 + lane) * 8;
  int mh = lane >> 4, nn = lane & 15;

  for (int t = 0; t < T_; ++t) {
    const unsigned* hp  = (t & 1) ? hpackB : hpackA;
    unsigned*       hpn = (t & 1) ? hpackA : hpackB;

    // ---- GEMM phase: gates_h tile = h @ Wh (K = 1024), software-pipelined ----
    v8f acc = {};
    const unsigned* ap = hp + lane * 8;
    const unsigned* bp = bbase;
    v16bf A = *(const v16bf*)ap;
    v16bf Bv = *(const v16bf*)bp;
#pragma unroll 4
    for (int kc = 0; kc < KC_ - 1; ++kc) {
      ap += 256; bp += 256;
      v16bf An = *(const v16bf*)ap;
      v16bf Bn = *(const v16bf*)bp;
      acc = __builtin_amdgcn_wmma_f32_16x16x32_bf16(false, A, false, Bv, (short)0, acc, false, false);
      A = An; Bv = Bn;
    }
    acc = __builtin_amdgcn_wmma_f32_16x16x32_bf16(false, A, false, Bv, (short)0, acc, false, false);

#pragma unroll
    for (int r = 0; r < 8; ++r)
      gates[(size_t)(r + 8 * mh) * G_ + wv * 16 + nn] = acc[r];

    grid_sync(bar_cnt, bar_gen, RBLK_, &lgen);

    // ---- pointwise LSTM cell: 16384 elements over 8192 threads ----
#pragma unroll
    for (int i = 0; i < 2; ++i) {
      int e = gtid + i * 8192;
      int m = e >> 10, j = e & 1023;
      size_t gb = (size_t)m * G_ + j;
      size_t xb = ((size_t)t * 16 + m) * G_ + j;
      float gi = gates[gb         ] + bf16_to_f32(xg[xb         ]);
      float gf = gates[gb + 1024  ] + bf16_to_f32(xg[xb + 1024  ]);
      float gg = gates[gb + 2048  ] + bf16_to_f32(xg[xb + 2048  ]);
      float go = gates[gb + 3072  ] + bf16_to_f32(xg[xb + 3072  ]);
      float c  = cstate[e];
      float cn = sigmoidf(gf) * c + sigmoidf(gi) * tanhf(gg);
      float hn = sigmoidf(go) * tanhf(cn);
      cstate[e] = cn;
      unsigned short hb = f32_to_bf16(hn);
      int hidx = hpack_ushort_index(m, j);
      ((unsigned short*)hpn)[hidx] = hb;                       // next step's A operand
      if (ypack)   ypack[(size_t)t * 16384 + hidx] = hb;       // layer0 -> layer1 input
      if (out_f32) out_f32[((size_t)m * T_ + t) * F_ + j] = hn;// layer1 -> d_out
      if (t == T_ - 1) { cfin[e] = cn; hfin[e] = hn; }
    }

    grid_sync(bar_cnt, bar_gen, RBLK_, &lgen);
  }
}

// ------------------------- host launch --------------------------------------
extern "C" void kernel_launch(void* const* d_in, const int* in_sizes, int n_in,
                              void* d_out, int out_size, void* d_ws, size_t ws_size,
                              hipStream_t stream) {
  (void)in_sizes; (void)n_in; (void)out_size; (void)ws_size;
  const float* x  = (const float*)d_in[0];   // [16,512,1024]
  const float* Wi = (const float*)d_in[1];   // [2,1024,4096]
  const float* Wh = (const float*)d_in[2];   // [2,1024,4096]
  const float* bb = (const float*)d_in[3];   // [2,4096]
  const float* c0 = (const float*)d_in[4];   // [2,16,1024]
  const float* h0 = (const float*)d_in[5];   // [2,16,1024]
  float* out = (float*)d_out;
  char* ws = (char*)d_ws;

  const size_t WPACK_B = (size_t)NT_ * KC_ * 32 * 8 * 4;   // 8 MB per weight
  const size_t XPACK_B = (size_t)T_ * KC_ * 32 * 8 * 4;    // 16 MB
  size_t off = 0;
  auto take = [&](size_t bytes) { size_t o = off; off += (bytes + 255) & ~(size_t)255; return o; };
  unsigned* WiP0 = (unsigned*)(ws + take(WPACK_B));
  unsigned* WiP1 = (unsigned*)(ws + take(WPACK_B));
  unsigned* WhP0 = (unsigned*)(ws + take(WPACK_B));
  unsigned* WhP1 = (unsigned*)(ws + take(WPACK_B));
  unsigned* xpack = (unsigned*)(ws + take(XPACK_B));
  unsigned* ypack = (unsigned*)(ws + take(XPACK_B));
  unsigned short* xg = (unsigned short*)(ws + take((size_t)T_ * 16 * G_ * 2)); // 64 MB
  float* gates  = (float*)(ws + take((size_t)16 * G_ * 4));
  float* cstate = (float*)(ws + take((size_t)16 * F_ * 4));
  unsigned* hpackA = (unsigned*)(ws + take((size_t)KC_ * 32 * 8 * 4));
  unsigned* hpackB = (unsigned*)(ws + take((size_t)KC_ * 32 * 8 * 4));
  unsigned* bar_cnt = (unsigned*)(ws + take(256));
  unsigned* bar_gen = bar_cnt + 1;

  const size_t OUT_C = (size_t)B_ * T_ * F_;       // 8,388,608
  const size_t OUT_H = OUT_C + 2 * 16384;

  // pack weights + input
  pack_w_kernel<<<8192, 256, 0, stream>>>(Wi,                 WiP0);
  pack_w_kernel<<<8192, 256, 0, stream>>>(Wi + (size_t)F_*G_, WiP1);
  pack_w_kernel<<<8192, 256, 0, stream>>>(Wh,                 WhP0);
  pack_w_kernel<<<8192, 256, 0, stream>>>(Wh + (size_t)F_*G_, WhP1);
  pack_x_kernel<<<16384, 256, 0, stream>>>(x, xpack);

  for (int l = 0; l < 2; ++l) {
    init_state_kernel<<<64, 256, 0, stream>>>(c0 + (size_t)l * 16384, h0 + (size_t)l * 16384,
                                              cstate, hpackA, bar_cnt, bar_gen);
    gemm_xg_kernel<<<4096, 256, 0, stream>>>(l == 0 ? xpack : ypack,
                                             l == 0 ? WiP0 : WiP1,
                                             bb + (size_t)l * G_, xg);
    lstm_recur_kernel<<<RBLK_, 256, 0, stream>>>(l == 0 ? WhP0 : WhP1, xg, gates,
                                                 hpackA, hpackB, cstate, bar_cnt, bar_gen,
                                                 l == 0 ? (unsigned short*)ypack : (unsigned short*)nullptr,
                                                 l == 1 ? out : (float*)nullptr,
                                                 out + OUT_C + (size_t)l * 16384,
                                                 out + OUT_H + (size_t)l * 16384);
  }
}
